// LowRankAttention_80453327388828
// MI455X (gfx1250) — matrix operand order repaired and checked by reading, not compiled, not run on hardware
//
#include <hip/hip_runtime.h>
#include <cstdint>
#include <cstddef>

// ---------------------------------------------------------------------------
// LowRankAttention on MI455X (gfx1250).
// rank(QK^T) <= D_HEAD=64 < svd_rank=102  =>  SVD reconstruction is exact
// =>  reference == plain softmax attention.  Implement FlashAttention with
// v_wmma_f32_16x16x32_f16 (wave32), online softmax via ds_swizzle cross-half
// reductions.
// ---------------------------------------------------------------------------

typedef __attribute__((ext_vector_type(16))) _Float16     v16h;
typedef __attribute__((ext_vector_type(2)))  _Float16     v2h;
typedef __attribute__((ext_vector_type(8)))  float        v8f;
typedef __attribute__((ext_vector_type(8)))  unsigned int v8u;

#define BATCH   2
#define NHEADS 16
#define SEQ    1024
#define DHEAD  64
// softmax scale (1/sqrt(64)) * log2(e), folded into Q during conversion so the
// main loop can use exp2 directly.
#define QSCALE 0.1803368801111953f

#define SWZ_XOR16 0x401F  // group-of-32 swizzle: and=0x1f, or=0, xor=0x10

// ---------------------------------------------------------------------------
// Pre-pass 1: Q (f32 [B,S,1024]) -> Qh (f16 [B,H,S,64]) * QSCALE.
// Per-head row-major == the contiguous-k B-operand order used here (lane = query col,
// 16 consecutive dh per lane-half).
// ---------------------------------------------------------------------------
__global__ __launch_bounds__(256) void lra_prep_q(const float* __restrict__ Q,
                                                  _Float16* __restrict__ Qh) {
  int idx = blockIdx.x * blockDim.x + threadIdx.x;       // 2^21 elements
  int d = idx & 63;
  int s = (idx >> 6) & 1023;
  int h = (idx >> 16) & 15;
  int b = idx >> 20;
  float v = Q[((size_t)b * SEQ + s) * (NHEADS * DHEAD) + h * DHEAD + d] * QSCALE;
  Qh[idx] = (_Float16)v;   // Qh[((b*16+h)*1024+s)*64+d] == Qh[idx]
}

// ---------------------------------------------------------------------------
// Pre-pass 2: K -> A-matrix fragments for the score GEMM (D[M=key,N=query]).
// Storage: [bh][t16=0..63][c=0..1][lane=0..31][i=0..15] f16, i.e. each lane
// loads 16 contiguous f16 (32B) per fragment.  A-layout (16-bit A 16x32):
//   M = lane&15 (key within tile), element i (v=i>>1,pos=i&1):
//   dh = 32*c + (lane>>4)*16 + i
// ---------------------------------------------------------------------------
__global__ __launch_bounds__(256) void lra_prep_k(const float* __restrict__ K,
                                                  _Float16* __restrict__ Kf) {
  int idx = blockIdx.x * blockDim.x + threadIdx.x;       // 2^21 elements
  int i    = idx & 15;
  int lane = (idx >> 4) & 31;
  int c    = (idx >> 9) & 1;
  int t    = (idx >> 10) & 63;
  int bh   = idx >> 16;
  int h = bh & 15, b = bh >> 4;
  int v = i >> 1, pos = i & 1;
  int dh  = 32 * c + (lane >> 4) * 16 + i;
  int key = t * 16 + (lane & 15);
  Kf[idx] = (_Float16)K[((size_t)b * SEQ + key) * (NHEADS * DHEAD) + h * DHEAD + dh];
}

// ---------------------------------------------------------------------------
// Pre-pass 3: V -> A-matrix fragments of V^T for the PV GEMM (D = O^T[d,q]).
// Storage: [bh][t32=0..31][c4=0..3][lane][i] f16.
//   M = d-in-chunk = lane&15  =>  d = 16*c4 + (lane&15)
//   K-dim = key-in-tile: k = 32*t32 + (lane>>4)*16 + i
// ---------------------------------------------------------------------------
__global__ __launch_bounds__(256) void lra_prep_v(const float* __restrict__ V,
                                                  _Float16* __restrict__ Vf) {
  int idx = blockIdx.x * blockDim.x + threadIdx.x;       // 2^21 elements
  int i    = idx & 15;
  int lane = (idx >> 4) & 31;
  int c4   = (idx >> 9) & 3;
  int t    = (idx >> 11) & 31;
  int bh   = idx >> 16;
  int h = bh & 15, b = bh >> 4;
  int v = i >> 1, pos = i & 1;
  int d = 16 * c4 + (lane & 15);
  int k = 32 * t + (lane >> 4) * 16 + i;
  Vf[idx] = (_Float16)V[((size_t)b * SEQ + k) * (NHEADS * DHEAD) + h * DHEAD + d];
}

// ---------------------------------------------------------------------------
// Main flash-attention kernel: one wave per (bh, 16-query block).
// 2048 waves total; 8 waves / 256-thread block.
// ---------------------------------------------------------------------------
__global__ __launch_bounds__(256) void lra_flash(const _Float16* __restrict__ Qh,
                                                 const _Float16* __restrict__ Kf,
                                                 const _Float16* __restrict__ Vf,
                                                 float* __restrict__ out) {
  const int lane = threadIdx.x & 31;
  const int wave = blockIdx.x * (blockDim.x >> 5) + (threadIdx.x >> 5);
  const int bh   = wave >> 6;          // 0..31  (b*16+h)
  const int qb   = wave & 63;          // 16-query block
  const int half = lane >> 4;
  const int ln   = lane & 15;

  // Q as WMMA-B operand: lane = query column, 16 contiguous dh per chunk-half.
  const _Float16* qptr = Qh + (((size_t)bh * SEQ) + qb * 16 + ln) * DHEAD + half * 16;
  const v16h qf0 = *(const v16h*)(qptr);        // dh chunk 0 (0..31)
  const v16h qf1 = *(const v16h*)(qptr + 32);   // dh chunk 1 (32..63)

  v8f o0 = {}, o1 = {}, o2 = {}, o3 = {};       // O^T accumulators, 4 d-chunks
  float m = -1e30f;                             // running max (log2 domain)
  float l = 0.0f;                               // running sum

  const _Float16* kbase = Kf + (size_t)bh * 65536 + lane * 16;
  const _Float16* vbase = Vf + (size_t)bh * 65536 + lane * 16;

  for (int t = 0; t < 32; ++t) {                // 32-key tiles
    const _Float16* kp = kbase + t * 2048;
    v16h ka0 = *(const v16h*)(kp);              // keys t*32+0..15, dh 0..31
    v16h ka1 = *(const v16h*)(kp + 512);        // keys t*32+0..15, dh 32..63
    v16h kb0 = *(const v16h*)(kp + 1024);       // keys t*32+16..31, dh 0..31
    v16h kb1 = *(const v16h*)(kp + 1536);

    // Scores (already scaled by 1/8*log2e via Q):  D[M=key, N=query]
    v8f d0 = {}, d1 = {};
    d0 = __builtin_amdgcn_wmma_f32_16x16x32_f16(false, ka0, false, qf0, (short)0, d0, false, false);
    d0 = __builtin_amdgcn_wmma_f32_16x16x32_f16(false, ka1, false, qf1, (short)0, d0, false, false);
    d1 = __builtin_amdgcn_wmma_f32_16x16x32_f16(false, kb0, false, qf0, (short)0, d1, false, false);
    d1 = __builtin_amdgcn_wmma_f32_16x16x32_f16(false, kb1, false, qf1, (short)0, d1, false, false);

    // Per-query max over the 32 keys of this tile: 8 in-lane rows of d0 + d1,
    // then one cross-half exchange (lane ^ 16 holds the other 8 keys).
    float tm = fmaxf(d0[0], d1[0]);
#pragma unroll
    for (int r = 1; r < 8; ++r) tm = fmaxf(tm, fmaxf(d0[r], d1[r]));
    tm = fmaxf(tm, __builtin_bit_cast(float,
           __builtin_amdgcn_ds_swizzle(__builtin_bit_cast(int, tm), SWZ_XOR16)));

    const float mnew = fmaxf(m, tm);
    const float sc   = __builtin_amdgcn_exp2f(m - mnew);   // rescale factor

    float e0[8], e1[8];
    float lsum = 0.0f;
#pragma unroll
    for (int r = 0; r < 8; ++r) {
      e0[r] = __builtin_amdgcn_exp2f(d0[r] - mnew);
      e1[r] = __builtin_amdgcn_exp2f(d1[r] - mnew);
      lsum += e0[r] + e1[r];
    }
    lsum += __builtin_bit_cast(float,
           __builtin_amdgcn_ds_swizzle(__builtin_bit_cast(int, lsum), SWZ_XOR16));
    l = l * sc + lsum;
    m = mnew;

#pragma unroll
    for (int r = 0; r < 8; ++r) { o0[r] *= sc; o1[r] *= sc; o2[r] *= sc; o3[r] *= sc; }

    // Build P^T as the WMMA-B operand (lane = query col; i -> k = 16*half+i).
    // Own-half values pack directly; the other 8 k's come from lane^16.
    unsigned int p0[4], p1[4], c0[4], c1[4];
#pragma unroll
    for (int j = 0; j < 4; ++j) {
      p0[j] = __builtin_bit_cast(unsigned int,
                __builtin_amdgcn_cvt_pkrtz(e0[2 * j], e0[2 * j + 1]));
      p1[j] = __builtin_bit_cast(unsigned int,
                __builtin_amdgcn_cvt_pkrtz(e1[2 * j], e1[2 * j + 1]));
      c0[j] = (unsigned int)__builtin_amdgcn_ds_swizzle((int)p0[j], SWZ_XOR16);
      c1[j] = (unsigned int)__builtin_amdgcn_ds_swizzle((int)p1[j], SWZ_XOR16);
    }
    v8u pu;
#pragma unroll
    for (int j = 0; j < 4; ++j) {
      pu[j]     = half ? c1[j] : p0[j];   // k = 16*half + 2j,2j+1
      pu[4 + j] = half ? p1[j] : c0[j];   // k = 16*half + 8 + 2j,2j+1
    }
    const v16h pf = __builtin_bit_cast(v16h, pu);

    // O^T += V^T * P^T   (4 d-chunks of 16)
    const _Float16* vp = vbase + t * 2048;
    v16h vf0 = *(const v16h*)(vp);
    v16h vf1 = *(const v16h*)(vp + 512);
    v16h vf2 = *(const v16h*)(vp + 1024);
    v16h vf3 = *(const v16h*)(vp + 1536);
    o0 = __builtin_amdgcn_wmma_f32_16x16x32_f16(false, vf0, false, pf, (short)0, o0, false, false);
    o1 = __builtin_amdgcn_wmma_f32_16x16x32_f16(false, vf1, false, pf, (short)0, o1, false, false);
    o2 = __builtin_amdgcn_wmma_f32_16x16x32_f16(false, vf2, false, pf, (short)0, o2, false, false);
    o3 = __builtin_amdgcn_wmma_f32_16x16x32_f16(false, vf3, false, pf, (short)0, o3, false, false);
  }

  const float inv = 1.0f / l;   // l is identical in both lane halves

  // O^T layout: lane holds query q = qb*16+ln; d = 16*c + 8*half + r.
  // Output: out[b, q, h*64 + d]  -> 8 contiguous floats per chunk per lane.
  float* ob = out + (((size_t)(bh >> 4) * SEQ) + qb * 16 + ln) * (NHEADS * DHEAD)
                  + (bh & 15) * DHEAD + half * 8;
  const v8f oc0 = o0, oc1 = o1, oc2 = o2, oc3 = o3;
  {
    float4 a, b2;
    a  = make_float4(oc0[0]*inv, oc0[1]*inv, oc0[2]*inv, oc0[3]*inv);
    b2 = make_float4(oc0[4]*inv, oc0[5]*inv, oc0[6]*inv, oc0[7]*inv);
    *(float4*)(ob +  0) = a;  *(float4*)(ob +  4) = b2;
    a  = make_float4(oc1[0]*inv, oc1[1]*inv, oc1[2]*inv, oc1[3]*inv);
    b2 = make_float4(oc1[4]*inv, oc1[5]*inv, oc1[6]*inv, oc1[7]*inv);
    *(float4*)(ob + 16) = a;  *(float4*)(ob + 20) = b2;
    a  = make_float4(oc2[0]*inv, oc2[1]*inv, oc2[2]*inv, oc2[3]*inv);
    b2 = make_float4(oc2[4]*inv, oc2[5]*inv, oc2[6]*inv, oc2[7]*inv);
    *(float4*)(ob + 32) = a;  *(float4*)(ob + 36) = b2;
    a  = make_float4(oc3[0]*inv, oc3[1]*inv, oc3[2]*inv, oc3[3]*inv);
    b2 = make_float4(oc3[4]*inv, oc3[5]*inv, oc3[6]*inv, oc3[7]*inv);
    *(float4*)(ob + 48) = a;  *(float4*)(ob + 52) = b2;
  }
}

// ---------------------------------------------------------------------------
extern "C" void kernel_launch(void* const* d_in, const int* in_sizes, int n_in,
                              void* d_out, int out_size, void* d_ws, size_t ws_size,
                              hipStream_t stream) {
  const float* Q = (const float*)d_in[0];
  const float* K = (const float*)d_in[1];
  const float* V = (const float*)d_in[2];
  float* out = (float*)d_out;

  const size_t per = (size_t)BATCH * NHEADS * SEQ * DHEAD;  // 2^21 f16 elems
  _Float16* Qh = (_Float16*)d_ws;            // 4 MB
  _Float16* Kf = Qh + per;                   // 4 MB
  _Float16* Vf = Kf + per;                   // 4 MB  (12 MB workspace total)

  const int n = (int)per;                    // 2,097,152
  lra_prep_q<<<n / 256, 256, 0, stream>>>(Q, Qh);
  lra_prep_k<<<n / 256, 256, 0, stream>>>(K, Kf);
  lra_prep_v<<<n / 256, 256, 0, stream>>>(V, Vf);

  // 32 (B*H) * 64 q-blocks = 2048 waves = 65536 threads
  lra_flash<<<(BATCH * NHEADS * (SEQ / 16) * 32) / 256, 256, 0, stream>>>(Qh, Kf, Vf, out);
}
